// MLA_50062138802426
// MI455X (gfx1250) — compile-verified, matrix operand order checked by reading, not run on hardware
//
#include <hip/hip_runtime.h>
#include <hip/hip_bf16.h>
#include <cmath>

// ---------------------------------------------------------------------------
// MLA forward for MI455X (gfx1250): bf16 WMMA everywhere, flash attention.
// ---------------------------------------------------------------------------

typedef __attribute__((ext_vector_type(16))) __bf16 v16bf;
typedef __attribute__((ext_vector_type(8)))  float  v8f;

#define DIM     2048
#define N_HEADS 16
#define KV_LORA 512
#define NOPE    128
#define ROPE    64
#define V_DIM   128
#define QK_DIM  192
#define BATCH   2
#define SEQ     2048
#define NTOK    (BATCH * SEQ)

// Optional gfx1250 async LDS copy path (probe via __has_builtin; safe fallback)
#if defined(__has_builtin)
#if __has_builtin(__builtin_amdgcn_global_load_async_to_lds_b128) && \
    __has_builtin(__builtin_amdgcn_s_wait_asynccnt)
#define HAVE_ASYNC_LDS 1
#endif
#endif

#ifdef HAVE_ASYNC_LDS
// exact param types per hipcc diagnostic: (int4 __device__*, int4 __shared__*, Ii, Ii)
typedef int async_i4 __attribute__((vector_size(16)));
typedef __attribute__((address_space(1))) async_i4 g_i4;
typedef __attribute__((address_space(3))) async_i4 l_i4;
#endif

union AFrag { v16bf v; unsigned int u[8]; };
union OFrag { v8f   v; float        f[8]; };

__device__ __forceinline__ unsigned short f32_to_bf16(float f) {
  unsigned int u = __float_as_uint(f);
  u = (u + 0x7FFFu + ((u >> 16) & 1u)) >> 16;
  return (unsigned short)u;
}

// 16-bit A/B operand: dword j of lane (h = lane>>4) holds K pair starting at:
__device__ __forceinline__ int kpair_off(int j, int h) {
  return (j < 4 ? 2 * j : 16 + 2 * (j - 4)) + 8 * h;
}

__device__ __forceinline__ v8f wmma_bf16(const AFrag& a, const AFrag& b, v8f c) {
  return __builtin_amdgcn_wmma_f32_16x16x32_bf16(
      /*neg_a=*/false, a.v, /*neg_b=*/false, b.v,
      /*c_mod=*/(short)0, c, /*reuse_a=*/false, /*reuse_b=*/false);
}

// ---------------------------------------------------------------------------
// Generic GEMM: C[M][N] = A[M][K] @ B[N][K]^T   (A,B,C f32; math in bf16 WMMA)
// Block tile 128x64, K-slab 64, 256 threads (8 waves), wave tile 32x32.
// Per staged slab each wave issues 8 WMMAs with A-fragment reuse.
// ---------------------------------------------------------------------------
#define BM 128
#define BN 64
#define BK 64

__global__ __launch_bounds__(256) void gemm_bf16w(
    const float* __restrict__ A, const float* __restrict__ B,
    float* __restrict__ C, int M, int N, int K)
{
  __shared__ unsigned short As[BM][BK];   // 16 KB
  __shared__ unsigned short Bs[BN][BK];   // 8 KB

  const int bm = blockIdx.y * BM;
  const int bn = blockIdx.x * BN;
  const int tid  = threadIdx.x;
  const int wave = tid >> 5;
  const int lane = tid & 31;
  const int wm = wave >> 1;        // 0..3  (32-row stripe)
  const int wn = wave & 1;         // 0..1  (32-col stripe)
  const int ln = lane & 15;
  const int lh = lane >> 4;

  OFrag acc[2][2];
#pragma unroll
  for (int i = 0; i < 2; ++i)
#pragma unroll
    for (int j = 0; j < 2; ++j) acc[i][j].v = (v8f){};

  for (int k0 = 0; k0 < K; k0 += BK) {
    // stage A slab: 128x64 f32 -> bf16, float4 (b128) granularity
    for (int i = tid * 4; i < BM * BK; i += 1024) {
      int r = i >> 6, c = i & 63;
      float4 v = *(const float4*)&A[(bm + r) * K + k0 + c];
      As[r][c]     = f32_to_bf16(v.x);
      As[r][c + 1] = f32_to_bf16(v.y);
      As[r][c + 2] = f32_to_bf16(v.z);
      As[r][c + 3] = f32_to_bf16(v.w);
    }
    // stage B slab: 64x64
    for (int i = tid * 4; i < BN * BK; i += 1024) {
      int r = i >> 6, c = i & 63;
      float4 v = *(const float4*)&B[(bn + r) * K + k0 + c];
      Bs[r][c]     = f32_to_bf16(v.x);
      Bs[r][c + 1] = f32_to_bf16(v.y);
      Bs[r][c + 2] = f32_to_bf16(v.z);
      Bs[r][c + 3] = f32_to_bf16(v.w);
    }
    // prefetch next slab into cache (global_prefetch_b8)
    if (k0 + BK < K) {
      __builtin_prefetch(&A[(bm + (tid >> 1)) * K + k0 + BK + (tid & 1) * 32], 0, 1);
      if (tid < 128)
        __builtin_prefetch(&B[(bn + (tid & 63)) * K + k0 + BK + (tid >> 6) * 32], 0, 1);
    }
    __syncthreads();

#pragma unroll
    for (int kk = 0; kk < BK; kk += 32) {
      AFrag a0, a1, b0, b1;
#pragma unroll
      for (int j = 0; j < 8; ++j) {
        int off = kk + kpair_off(j, lh);
        a0.u[j] = *(const unsigned int*)&As[wm * 32 + ln][off];
        a1.u[j] = *(const unsigned int*)&As[wm * 32 + 16 + ln][off];
        b0.u[j] = *(const unsigned int*)&Bs[wn * 32 + ln][off];
        b1.u[j] = *(const unsigned int*)&Bs[wn * 32 + 16 + ln][off];
      }
      acc[0][0].v = wmma_bf16(a0, b0, acc[0][0].v);
      acc[0][1].v = wmma_bf16(a0, b1, acc[0][1].v);
      acc[1][0].v = wmma_bf16(a1, b0, acc[1][0].v);
      acc[1][1].v = wmma_bf16(a1, b1, acc[1][1].v);
    }
    __syncthreads();
  }

  const int cbase = bn + wn * 32 + ln;
#pragma unroll
  for (int mi = 0; mi < 2; ++mi)
#pragma unroll
    for (int r = 0; r < 8; ++r) {
      int row = bm + wm * 32 + mi * 16 + r + 8 * lh;
      C[row * N + cbase]      = acc[mi][0].f[r];
      C[row * N + cbase + 16] = acc[mi][1].f[r];
    }
}

// ---------------------------------------------------------------------------
// Per-token postprocess: RMSNorm(kv_c), RoPE(k_pe), RoPE(q_pe)+scale, q->bf16.
// ---------------------------------------------------------------------------
__global__ __launch_bounds__(256) void postproc(
    const float* __restrict__ q_raw,     // [NTOK][H*QK_DIM]
    const float* __restrict__ kv_raw,    // [NTOK][KV_LORA+ROPE]
    const float* __restrict__ kv_norm_w, // [KV_LORA]
    const float* __restrict__ fcos,      // [SEQ][ROPE/2]
    const float* __restrict__ fsin,      // [SEQ][ROPE/2]
    float scale,
    float* __restrict__ kv_c_norm,       // [NTOK][KV_LORA]
    float* __restrict__ k_pe_rot,        // [NTOK][ROPE]
    unsigned short* __restrict__ q_bf)   // [B][H][S][QK_DIM] bf16
{
  const int tok = blockIdx.x;
  const int s = tok & (SEQ - 1);
  const int b = tok >> 11;
  const int tid = threadIdx.x;
  __shared__ float red[256];

  const float* kvr = kv_raw + tok * (KV_LORA + ROPE);

  float ss = 0.f;
  for (int i = tid; i < KV_LORA; i += 256) { float v = kvr[i]; ss += v * v; }
  red[tid] = ss;
  __syncthreads();
  for (int off = 128; off > 0; off >>= 1) {
    if (tid < off) red[tid] += red[tid + off];
    __syncthreads();
  }
  const float inv = rsqrtf(red[0] / (float)KV_LORA + 1e-6f);

  for (int i = tid; i < KV_LORA; i += 256)
    kv_c_norm[tok * KV_LORA + i] = kvr[i] * inv * kv_norm_w[i];

  if (tid < ROPE / 2) {
    float c  = fcos[s * (ROPE / 2) + tid];
    float sn = fsin[s * (ROPE / 2) + tid];
    float re = kvr[KV_LORA + 2 * tid];
    float im = kvr[KV_LORA + 2 * tid + 1];
    k_pe_rot[tok * ROPE + 2 * tid]     = re * c - im * sn;
    k_pe_rot[tok * ROPE + 2 * tid + 1] = re * sn + im * c;
  }

  const float* qr = q_raw + tok * (N_HEADS * QK_DIM);
  for (int i = tid; i < N_HEADS * QK_DIM; i += 256) {
    int hh = i / QK_DIM, d = i % QK_DIM;
    float val;
    if (d < NOPE) {
      val = qr[hh * QK_DIM + d];
    } else {
      int p = (d - NOPE) >> 1;
      float c  = fcos[s * (ROPE / 2) + p];
      float sn = fsin[s * (ROPE / 2) + p];
      float re = qr[hh * QK_DIM + NOPE + 2 * p];
      float im = qr[hh * QK_DIM + NOPE + 2 * p + 1];
      val = ((d & 1) == 0) ? (re * c - im * sn) : (re * sn + im * c);
    }
    q_bf[((b * N_HEADS + hh) * SEQ + s) * QK_DIM + d] = f32_to_bf16(val * scale);
  }
}

// ---------------------------------------------------------------------------
// Split kvb -> bf16 K (nope || broadcast rope) and bf16 V, [B][H][S][*] layout
// ---------------------------------------------------------------------------
__global__ __launch_bounds__(256) void buildkv(
    const float* __restrict__ kvb,      // [NTOK][H*(NOPE+V_DIM)]
    const float* __restrict__ k_pe,     // [NTOK][ROPE]
    unsigned short* __restrict__ k_bf,  // [B][H][S][QK_DIM]
    unsigned short* __restrict__ v_bf)  // [B][H][S][V_DIM]
{
  const int tok = blockIdx.x;
  const int s = tok & (SEQ - 1);
  const int b = tok >> 11;
  const int tid = threadIdx.x;
  const float* row = kvb + tok * (N_HEADS * (NOPE + V_DIM));

  for (int i = tid; i < N_HEADS * (NOPE + V_DIM); i += 256) {
    int h = i >> 8, d = i & 255;
    float v = row[h * (NOPE + V_DIM) + d];
    if (d < NOPE)
      k_bf[((b * N_HEADS + h) * SEQ + s) * QK_DIM + d] = f32_to_bf16(v);
    else
      v_bf[((b * N_HEADS + h) * SEQ + s) * V_DIM + (d - NOPE)] = f32_to_bf16(v);
  }
  for (int i = tid; i < N_HEADS * ROPE; i += 256) {
    int h = i >> 6, d = i & 63;
    k_bf[((b * N_HEADS + h) * SEQ + s) * QK_DIM + NOPE + d] =
        f32_to_bf16(k_pe[tok * ROPE + d]);
  }
}

// ---------------------------------------------------------------------------
// Flash attention: 64 queries / workgroup, 4 waves x 16 query rows each.
// K block async-copied to LDS (ASYNCcnt path) when available; online softmax
// via 16-lane shfl reductions matching the 16x16 C-matrix VGPR layout.
// ---------------------------------------------------------------------------
__global__ __launch_bounds__(128) void mla_attn(
    const unsigned short* __restrict__ q_bf, // [B][H][S][QK_DIM]
    const unsigned short* __restrict__ k_bf, // [B][H][S][QK_DIM]
    const unsigned short* __restrict__ v_bf, // [B][H][S][V_DIM]
    float* __restrict__ attn)                // [NTOK][H*V_DIM] f32
{
  __shared__ __align__(16) unsigned short Ks[32][QK_DIM]; // key block row-major
  __shared__ unsigned short Vt[V_DIM][32];                // value block transposed
  __shared__ unsigned short Ps[4][16][32];                // per-wave probs

  const int qb0 = blockIdx.x * 64;
  const int h = blockIdx.y, b = blockIdx.z;
  const int tid  = threadIdx.x;
  const int wave = tid >> 5;
  const int lane = tid & 31;
  const int ln = lane & 15, lh = lane >> 4;

  const int bh = b * N_HEADS + h;
  const unsigned short* qbase = q_bf + bh * (SEQ * QK_DIM);
  const unsigned short* kbase = k_bf + bh * (SEQ * QK_DIM);
  const unsigned short* vbase = v_bf + bh * (SEQ * V_DIM);

  // Q fragments for this wave's 16 rows (held in registers for whole loop)
  AFrag qf[6];
  const int qrow = qb0 + wave * 16 + ln;
#pragma unroll
  for (int c = 0; c < 6; ++c)
#pragma unroll
    for (int j = 0; j < 8; ++j)
      qf[c].u[j] = *(const unsigned int*)
          &qbase[qrow * QK_DIM + c * 32 + kpair_off(j, lh)];

  OFrag o[8];
#pragma unroll
  for (int t = 0; t < 8; ++t) o[t].v = (v8f){};
  float mrow[8], lrow[8];
#pragma unroll
  for (int r = 0; r < 8; ++r) { mrow[r] = -3.0e38f; lrow[r] = 0.f; }

  const int nkb = (qb0 + 64) / 32;  // causal: key blocks covering keys <= qb0+63
  for (int kb = 0; kb < nkb; ++kb) {
    const int k0 = kb * 32;
    __syncthreads();  // protect LDS reuse across iterations

#ifdef HAVE_ASYNC_LDS
    // stage K block 32x192 bf16 (12 KB) via async b128 copies: 768 chunks,
    // 6 per lane; completion via ASYNCcnt then workgroup barrier.
    {
#pragma unroll
      for (int it = 0; it < 6; ++it) {
        int idx = it * 128 + tid;          // 16-byte chunk id
        int r = idx / 24, c = idx % 24;    // key row, chunk within 384B row
        const unsigned short* gp = &kbase[(k0 + r) * QK_DIM + c * 8];
        unsigned short* lp = &Ks[r][c * 8];
        __builtin_amdgcn_global_load_async_to_lds_b128(
            (g_i4*)gp, (l_i4*)lp, 0, 0);
      }
      __builtin_amdgcn_s_wait_asynccnt(0);
    }
#else
    for (int i = tid; i < 32 * (QK_DIM / 2); i += 128) {
      int r = i / (QK_DIM / 2), c = i % (QK_DIM / 2);
      *((unsigned int*)&Ks[0][0] + r * (QK_DIM / 2) + c) =
          *(const unsigned int*)&kbase[(k0 + r) * QK_DIM + c * 2];
    }
#endif
    // stage V block transposed (keys become the fast axis for B-operand pairs)
    for (int i = tid; i < 32 * V_DIM; i += 128) {
      int r = i >> 7, d = i & (V_DIM - 1);
      Vt[d][r] = vbase[(k0 + r) * V_DIM + d];
    }
    __syncthreads();

    // scores: 16 queries x 32 keys = two 16x16 C tiles, K-depth 192 = 6 steps
    OFrag sc0, sc1;
    sc0.v = (v8f){}; sc1.v = (v8f){};
    AFrag kf;
#pragma unroll
    for (int c = 0; c < 6; ++c) {
#pragma unroll
      for (int j = 0; j < 8; ++j)
        kf.u[j] = *(const unsigned int*)&Ks[ln][c * 32 + kpair_off(j, lh)];
      sc0.v = wmma_bf16(qf[c], kf, sc0.v);
    }
#pragma unroll
    for (int c = 0; c < 6; ++c) {
#pragma unroll
      for (int j = 0; j < 8; ++j)
        kf.u[j] = *(const unsigned int*)&Ks[16 + ln][c * 32 + kpair_off(j, lh)];
      sc1.v = wmma_bf16(qf[c], kf, sc1.v);
    }

    // online softmax per query row (row = r + 8*lh lives in one 16-lane half)
#pragma unroll
    for (int r = 0; r < 8; ++r) {
      const int qi = qb0 + wave * 16 + r + 8 * lh;
      float s0 = (k0 + ln      <= qi) ? sc0.f[r] : -1e30f;
      float s1 = (k0 + 16 + ln <= qi) ? sc1.f[r] : -1e30f;
      float mx = fmaxf(s0, s1);
#pragma unroll
      for (int d = 1; d < 16; d <<= 1) mx = fmaxf(mx, __shfl_xor(mx, d, 32));
      const float nm = fmaxf(mrow[r], mx);
      const float corr = __expf(mrow[r] - nm);
      mrow[r] = nm;
      const float p0 = __expf(s0 - nm);
      const float p1 = __expf(s1 - nm);
      float rs = p0 + p1;
#pragma unroll
      for (int d = 1; d < 16; d <<= 1) rs += __shfl_xor(rs, d, 32);
      lrow[r] = lrow[r] * corr + rs;
#pragma unroll
      for (int t = 0; t < 8; ++t) o[t].f[r] *= corr;
      Ps[wave][r + 8 * lh][ln]      = f32_to_bf16(p0);
      Ps[wave][r + 8 * lh][16 + ln] = f32_to_bf16(p1);
    }
    __syncthreads();

    // O += P @ V : A = P (16x32), 8 output column tiles of 16
    AFrag pf;
#pragma unroll
    for (int j = 0; j < 8; ++j)
      pf.u[j] = *(const unsigned int*)&Ps[wave][ln][kpair_off(j, lh)];
#pragma unroll
    for (int t = 0; t < 8; ++t) {
      AFrag vf;
#pragma unroll
      for (int j = 0; j < 8; ++j)
        vf.u[j] = *(const unsigned int*)&Vt[t * 16 + ln][kpair_off(j, lh)];
      o[t].v = wmma_bf16(pf, vf, o[t].v);
    }
  }

  // normalize and write attn output (f32, row = token, col = h*128+d)
#pragma unroll
  for (int r = 0; r < 8; ++r) {
    const int qi = qb0 + wave * 16 + r + 8 * lh;
    const float invl = 1.0f / lrow[r];
    const int rowoff = (b * SEQ + qi) * (N_HEADS * V_DIM) + h * V_DIM;
#pragma unroll
    for (int t = 0; t < 8; ++t)
      attn[rowoff + t * 16 + ln] = o[t].f[r] * invl;
  }
}

// ---------------------------------------------------------------------------
// Host orchestration
// ---------------------------------------------------------------------------
extern "C" void kernel_launch(void* const* d_in, const int* in_sizes, int n_in,
                              void* d_out, int out_size, void* d_ws, size_t ws_size,
                              hipStream_t stream) {
  const float* x         = (const float*)d_in[0];
  const float* wq        = (const float*)d_in[1];
  const float* wkv_a     = (const float*)d_in[2];
  const float* kv_norm_w = (const float*)d_in[3];
  const float* wkv_b     = (const float*)d_in[4];
  const float* wo        = (const float*)d_in[5];
  const float* fcos      = (const float*)d_in[6];
  const float* fsin      = (const float*)d_in[7];
  float* out = (float*)d_out;

  // workspace carve-up
  char* ws = (char*)d_ws;
  size_t off = 0;
  auto take = [&](size_t bytes) { char* p = ws + off; off += (bytes + 255) & ~(size_t)255; return p; };
  float* q_raw   = (float*)take((size_t)NTOK * N_HEADS * QK_DIM * 4);
  float* kv_raw  = (float*)take((size_t)NTOK * (KV_LORA + ROPE) * 4);
  float* kv_c    = (float*)take((size_t)NTOK * KV_LORA * 4);
  float* k_pe    = (float*)take((size_t)NTOK * ROPE * 4);
  unsigned short* q_bf = (unsigned short*)take((size_t)NTOK * N_HEADS * QK_DIM * 2);
  float* kvb     = (float*)take((size_t)NTOK * N_HEADS * (NOPE + V_DIM) * 4);
  unsigned short* k_bf = (unsigned short*)take((size_t)NTOK * N_HEADS * QK_DIM * 2);
  unsigned short* v_bf = (unsigned short*)take((size_t)NTOK * N_HEADS * V_DIM * 2);
  float* attn    = (float*)take((size_t)NTOK * N_HEADS * V_DIM * 4);

  // softmax scale with yarn mscale (MAX_LEN > ORIG_LEN)
  const float m = 0.1f * 1.0f * logf(40.0f) + 1.0f;
  const float scale = (1.0f / sqrtf((float)QK_DIM)) * m * m;

  dim3 blk(256);
  // 1) q_raw = x @ wq^T   (4096 x 3072 x 2048)
  gemm_bf16w<<<dim3((N_HEADS * QK_DIM) / BN, NTOK / BM), blk, 0, stream>>>(
      x, wq, q_raw, NTOK, N_HEADS * QK_DIM, DIM);
  // 2) kv_raw = x @ wkv_a^T   (4096 x 576 x 2048)
  gemm_bf16w<<<dim3((KV_LORA + ROPE) / BN, NTOK / BM), blk, 0, stream>>>(
      x, wkv_a, kv_raw, NTOK, KV_LORA + ROPE, DIM);
  // 3) rmsnorm + rope + q->bf16
  postproc<<<dim3(NTOK), blk, 0, stream>>>(
      q_raw, kv_raw, kv_norm_w, fcos, fsin, scale, kv_c, k_pe, q_bf);
  // 4) kvb = kv_c @ wkv_b^T   (4096 x 4096 x 512)
  gemm_bf16w<<<dim3((N_HEADS * (NOPE + V_DIM)) / BN, NTOK / BM), blk, 0, stream>>>(
      kv_c, wkv_b, kvb, NTOK, N_HEADS * (NOPE + V_DIM), KV_LORA);
  // 5) split into K/V bf16
  buildkv<<<dim3(NTOK), blk, 0, stream>>>(kvb, k_pe, k_bf, v_bf);
  // 6) flash attention
  mla_attn<<<dim3(SEQ / 64, N_HEADS, BATCH), dim3(128), 0, stream>>>(
      q_bf, k_bf, v_bf, attn);
  // 7) out = attn @ wo^T   (4096 x 2048 x 2048)
  gemm_bf16w<<<dim3(DIM / BN, NTOK / BM), blk, 0, stream>>>(
      attn, wo, out, NTOK, DIM, DIM);
}
